// SpikeFP8Linear_MultiPrecision_11450382811509
// MI455X (gfx1250) — matrix operand order repaired and checked by reading, not compiled, not run on hardware
//
#include <hip/hip_runtime.h>
#include <hip/hip_bf16.h>
#include <cstdint>

typedef __attribute__((ext_vector_type(16))) _Float16 v16h;
typedef __attribute__((ext_vector_type(8)))  _Float16 v8h;
typedef __attribute__((ext_vector_type(8)))  float    v8f;

#define M_TOT 2048   // 8 * 256
#define N_TOT 1024   // out_features
#define K_TOT 1024   // in_features
#define NX (M_TOT * K_TOT)   // decoded activation count
#define NW (N_TOT * K_TOT)   // decoded weight count

// ---------------------------------------------------------------------------
// Phase 1: decode FP8 bit-pulses (8 floats/value, [s e3..e0 m2..m0], E4M3
// bias 7, subnormal at ev==0, ev==15 decoded as finite per the reference)
// into exact FP16 values. 2 x float4 loads per value, one f16 store.
// tid range is exactly NX+NW = 3,145,728 = 12288 * 256 -> no bounds check.
// ---------------------------------------------------------------------------
__global__ __launch_bounds__(256) void decode_fp8_kernel(
    const float* __restrict__ x, const float* __restrict__ w,
    _Float16* __restrict__ dst) {
  uint32_t tid = blockIdx.x * 256u + threadIdx.x;
  const float* src = (tid < NX) ? (x + (size_t)tid * 8)
                                : (w + (size_t)(tid - (uint32_t)NX) * 8);
  float4 b0 = ((const float4*)src)[0];
  float4 b1 = ((const float4*)src)[1];
  int s  = (b0.x != 0.0f);
  int ev = ((b0.y != 0.0f) << 3) | ((b0.z != 0.0f) << 2) |
           ((b0.w != 0.0f) << 1) |  (b1.x != 0.0f);
  int mn = ((b1.y != 0.0f) << 2) | ((b1.z != 0.0f) << 1) | (b1.w != 0.0f);
  float mag;
  if (ev == 0) {
    mag = (float)mn * 0x1p-9f;                      // (mn/8) * 2^-6
  } else {
    // (1 + mn/8) * 2^(ev-7) = (8+mn) * 2^(ev-10); exponent bits = ev-10+127
    mag = (float)(8 + mn) * __uint_as_float((uint32_t)(ev + 117) << 23);
  }
  float val = s ? -mag : mag;
  dst[tid] = (_Float16)val;                          // exact conversion
}

// ---------------------------------------------------------------------------
// Phase 2: fused WMMA GEMM (f16 inputs, f32 accumulate == reference math)
// plus FP32->bit-pulse encoding epilogue written as b128 stores.
// One wave per 16x16 output tile; 128x64 tiles; 8 waves per block.
// ---------------------------------------------------------------------------
__global__ __launch_bounds__(256) void gemm_encode_kernel(
    const _Float16* __restrict__ A,   // [M_TOT][K_TOT] decoded activations
    const _Float16* __restrict__ W,   // [N_TOT][K_TOT] decoded weights
    float* __restrict__ Out) {        // [M_TOT][N_TOT][32] bit pulses
  const uint32_t lane  = threadIdx.x & 31u;
  const uint32_t wave  = threadIdx.x >> 5;
  const uint32_t tile  = blockIdx.x * 8u + wave;   // 0..8191
  const uint32_t mTile = tile >> 6;                // 0..127
  const uint32_t nTile = tile & 63u;               // 0..63

  const uint32_t lh = lane >> 4;                   // half-wave: 0 or 1
  const uint32_t ll = lane & 15u;                  // row/col within tile

  // A fragment: lane holds row (mTile*16+ll); lh=0 -> K 0..15,
  // lh=1 -> K 16..31 (same K order as B).
  const _Float16* aRow = A + (size_t)(mTile * 16 + ll) * K_TOT;
  // B fragment: lane holds column (nTile*16+ll) = contiguous row of W;
  // lh=0 -> K 0..15, lh=1 -> K 16..31 (same K order as A).
  const _Float16* bRow = W + (size_t)(nTile * 16 + ll) * K_TOT;

  v8f c = {};
#pragma unroll 4
  for (uint32_t k = 0; k < K_TOT; k += 32) {
    v8h a0 = *(const v8h*)(aRow + k + lh * 16);
    v8h a1 = *(const v8h*)(aRow + k + lh * 16 + 8);
    v8h b0 = *(const v8h*)(bRow + k + lh * 16);
    v8h b1 = *(const v8h*)(bRow + k + lh * 16 + 8);
    v16h av = __builtin_shufflevector(a0, a1, 0, 1, 2, 3, 4, 5, 6, 7,
                                      8, 9, 10, 11, 12, 13, 14, 15);
    v16h bv = __builtin_shufflevector(b0, b1, 0, 1, 2, 3, 4, 5, 6, 7,
                                      8, 9, 10, 11, 12, 13, 14, 15);
    c = __builtin_amdgcn_wmma_f32_16x16x32_f16(
        /*neg_a=*/false, av, /*neg_b=*/false, bv,
        /*c_mod=*/(short)0, c, /*reuse_a=*/false, /*reuse_b=*/false);
  }

  // Epilogue: each accumulator element -> 32 bit pulses (sign/MSB first),
  // emitted as 8 float4 (b128) stores per element, 128B-aligned streams.
  const uint32_t o = nTile * 16 + ll;
#pragma unroll
  for (int r = 0; r < 8; ++r) {
    uint32_t m = mTile * 16 + lh * 8 + (uint32_t)r;  // C/D row mapping
    uint32_t bits = __float_as_uint(c[r]);
    float4* p = (float4*)(Out + ((size_t)m * N_TOT + o) * 32);
#pragma unroll
    for (int j = 0; j < 8; ++j) {
      int sh = 31 - 4 * j;
      float4 v;
      v.x = (float)((bits >> sh) & 1u);
      v.y = (float)((bits >> (sh - 1)) & 1u);
      v.z = (float)((bits >> (sh - 2)) & 1u);
      v.w = (float)((bits >> (sh - 3)) & 1u);
      p[j] = v;
    }
  }
}

extern "C" void kernel_launch(void* const* d_in, const int* in_sizes, int n_in,
                              void* d_out, int out_size, void* d_ws, size_t ws_size,
                              hipStream_t stream) {
  const float* x = (const float*)d_in[0];        // [8,256,1024,8] pulses
  const float* w = (const float*)d_in[1];        // [1024,1024,8] pulses
  float* out = (float*)d_out;                    // [8,256,1024,32] pulses
  _Float16* ws = (_Float16*)d_ws;                // 6 MB: A fp16 then W fp16

  // Phase 1: decode both operand tensors into fp16 workspace.
  decode_fp8_kernel<<<(NX + NW) / 256, 256, 0, stream>>>(x, w, ws);

  // Phase 2: WMMA GEMM + bit-pulse encode. 8192 tiles / 8 waves per block.
  gemm_encode_kernel<<<(M_TOT / 16) * (N_TOT / 16) / 8, 256, 0, stream>>>(
      ws, ws + (size_t)NX, out);
}
